// VariationalGIB_89026082111545
// MI455X (gfx1250) — compile-verified, matrix operand order checked by reading, not run on hardware
//
#include <hip/hip_runtime.h>
#include <hip/hip_bf16.h>
#include <math.h>

#define HD   128   // feature dim
#define NPG  32    // nodes per graph
#define EPG  128   // edges per graph

typedef __attribute__((ext_vector_type(16))) _Float16 v16h;
typedef __attribute__((ext_vector_type(8)))  _Float16 v8h;
typedef __attribute__((ext_vector_type(8)))  float    v8f;

// ---------------------------------------------------------------------------
// Prep kernel 1: convert W1 (128x128 f32, row-major [k][n]) into WMMA B-fragment
// layout in f16:  fragment index f = (ntile*4 + ktile)*32 + lane, 16 halves each.
// For lane L: n = ntile*16 + (L&15); klo = (L>=16)?8:0;
//   halves 0..7  -> K = ktile*32 + klo + e
//   halves 8..15 -> K = ktile*32 + 16 + klo + e
// ---------------------------------------------------------------------------
__global__ void pack_w1_kernel(const float* __restrict__ W1, _Float16* __restrict__ w1p) {
    int t = blockIdx.x * blockDim.x + threadIdx.x;
    if (t >= 8 * 4 * 32) return;
    int lane  = t & 31;
    int ktile = (t >> 5) & 3;
    int ntile = t >> 7;
    int n   = ntile * 16 + (lane & 15);
    int klo = (lane >= 16) ? 8 : 0;
    _Float16* dst = w1p + (size_t)t * 16;
#pragma unroll
    for (int e = 0; e < 8; ++e) {
        int k = ktile * 32 + klo + e;
        dst[e] = (_Float16)W1[k * HD + n];
    }
#pragma unroll
    for (int e = 0; e < 8; ++e) {
        int k = ktile * 32 + 16 + klo + e;
        dst[8 + e] = (_Float16)W1[k * HD + n];
    }
}

// Prep kernel 2: zero the three scalar accumulators (deterministic each call).
__global__ void zero_scalars_kernel(float* s) {
    int t = threadIdx.x;
    if (t < 3) s[t] = 0.0f;
}

// ---------------------------------------------------------------------------
// Fused per-graph kernel: 1 workgroup (128 threads = 4 waves) per graph.
// ---------------------------------------------------------------------------
__global__ __launch_bounds__(128)
void gib_fused_kernel(const float* __restrict__ x,
                      const _Float16* __restrict__ w1p,
                      const float* __restrict__ b1,
                      const float* __restrict__ W2,
                      const float* __restrict__ b2,
                      const float* __restrict__ gumbel,
                      const float* __restrict__ uniform,
                      const int*   __restrict__ esrc,
                      const int*   __restrict__ edst,
                      float* __restrict__ graph_emb,
                      float* __restrict__ noisy_emb,
                      float* __restrict__ active,
                      float* __restrict__ scalars,   // [pos_penalty, kl_loss, preserve_rate]
                      int G)
{
    __shared__ float    s_x[NPG][HD];     // 16 KB  f32 copy for stats/noisy
    __shared__ _Float16 s_xh[NPG][HD];    //  8 KB  f16 copy = WMMA A matrix
    __shared__ float    s_abs[NPG][HD];   // 16 KB  abstract1 = tanh(xW1+b1)
    __shared__ float    s_logit[NPG][2];
    __shared__ float    s_p0[NPG], s_p1[NPG], s_lp[NPG], s_ln[NPG];
    __shared__ float    s_red[128];
    __shared__ float    s_adj[4];
    __shared__ int      s_pres;

    const int g    = blockIdx.x;
    const int t    = threadIdx.x;      // 0..127
    const int lane = t & 31;
    const int wave = t >> 5;

    const float* xg = x + (size_t)g * NPG * HD;
    const float* ug = uniform + (size_t)g * NPG * HD;

    // Prefetch the uniform-noise tile (consumed after the GEMM) into cache.
#pragma unroll
    for (int i = 0; i < NPG; i += 8)
        __builtin_prefetch(&ug[i * HD + t], 0, 1);

    // Stage x into LDS (f32 + f16). Fully coalesced: thread t = feature column.
#pragma unroll
    for (int i = 0; i < NPG; ++i) {
        float v = xg[i * HD + t];
        s_x[i][t]  = v;
        s_xh[i][t] = (_Float16)v;
    }
    if (t == 0) { s_adj[0] = s_adj[1] = s_adj[2] = s_adj[3] = 0.0f; s_pres = 0; }
    __syncthreads();

    // ---------------- GEMM: abstract1 = tanh(x @ W1 + b1) via WMMA f16 -----
    // Output 32x128 = 2 Mtiles x 8 Ntiles. Wave w owns Ntiles {2w, 2w+1}.
    const int klo  = (lane >= 16) ? 8 : 0;
    const int mrow = lane & 15;
    v8f acc[2][2] = {};   // [mtile][nt]

#pragma unroll
    for (int kt = 0; kt < 4; ++kt) {
        v16h afrag[2];
#pragma unroll
        for (int mt = 0; mt < 2; ++mt) {
            const _Float16* ap = &s_xh[mt * 16 + mrow][kt * 32 + klo];
            v8h lo = *(const v8h*)ap;          // ds_load_b128
            v8h hi = *(const v8h*)(ap + 16);   // ds_load_b128
#pragma unroll
            for (int e = 0; e < 8; ++e) { afrag[mt][e] = lo[e]; afrag[mt][8 + e] = hi[e]; }
        }
#pragma unroll
        for (int nt = 0; nt < 2; ++nt) {
            int ntile = wave * 2 + nt;
            const _Float16* bp = w1p + ((size_t)((ntile * 4 + kt) * 32 + lane)) * 16;
            v16h bfrag = *(const v16h*)bp;     // one global_load_b128 pair, L2-hot
#pragma unroll
            for (int mt = 0; mt < 2; ++mt) {
                acc[mt][nt] = __builtin_amdgcn_wmma_f32_16x16x32_f16(
                    false, afrag[mt], false, bfrag,
                    (short)0, acc[mt][nt], false, false);
            }
        }
    }

    // bias + tanh, scatter C tiles to LDS per ISA C-layout:
    // VGPR r, lanes 0-15 -> M = r ; lanes 16-31 -> M = r + 8 ; N = lane&15.
    const int mhi = (lane >= 16) ? 8 : 0;
#pragma unroll
    for (int nt = 0; nt < 2; ++nt) {
        int ncol = (wave * 2 + nt) * 16 + (lane & 15);
        float bb = b1[ncol];
#pragma unroll
        for (int mt = 0; mt < 2; ++mt)
#pragma unroll
            for (int r = 0; r < 8; ++r)
                s_abs[mt * 16 + r + mhi][ncol] = tanhf(acc[mt][nt][r] + bb);
    }
    __syncthreads();

    // ---------------- head: logits = abstract1 @ W2 + b2 -------------------
    if (t < 2 * NPG) {
        int node = t >> 1, col = t & 1;
        float s = b2[col];
#pragma unroll 8
        for (int j = 0; j < HD; ++j) s += s_abs[node][j] * W2[j * 2 + col];
        s_logit[node][col] = s;
    }
    __syncthreads();

    // ---------------- per-node softmax, gumbel softmax, active -------------
    if (t < NPG) {
        float l0 = s_logit[t][0], l1 = s_logit[t][1];
        float m  = fmaxf(l0, l1);
        float e0 = __expf(l0 - m), e1 = __expf(l1 - m);
        float p0 = e0 / (e0 + e1), p1 = 1.0f - p0;
        const float* gn = gumbel + ((size_t)g * NPG + t) * 2;
        float g0 = p0 + gn[0], g1 = p1 + gn[1];
        float gm = fmaxf(g0, g1);
        float q0 = __expf(g0 - gm), q1 = __expf(g1 - gm);
        float lp = q0 / (q0 + q1), ln = 1.0f - lp;
        s_p0[t] = p0; s_p1[t] = p1; s_lp[t] = lp; s_ln[t] = ln;
        active[(size_t)g * NPG + t] = (lp > 0.5f) ? 1.0f : 0.0f;
        if (p0 > 0.5f) atomicAdd(&s_pres, 1);
    }
    __syncthreads();

    // ---------------- per-feature stats, noisy emb, KL ----------------------
    // thread t = feature j.
    float sum = 0.0f;
#pragma unroll
    for (int i = 0; i < NPG; ++i) sum += s_x[i][t];
    float mean = sum * (1.0f / NPG);
    float var = 0.0f;
#pragma unroll
    for (int i = 0; i < NPG; ++i) { float d = s_x[i][t] - mean; var += d * d; }
    var *= (1.0f / (NPG - 1));
    float stdv = sqrtf(var);
    float inv  = 1.0f / (stdv + 1e-7f);

    graph_emb[(size_t)g * HD + t] = sum;

    float nsum = 0.0f, t2 = 0.0f, S2 = 0.0f;
#pragma unroll
    for (int i = 0; i < NPG; ++i) {
        float lp = s_lp[i], ln = s_ln[i];
        float xv = s_x[i][t];
        float nm = lp * xv + ln * mean;         // noisy mean
        float ns = ln * stdv;                   // noisy std
        nsum += nm + ug[i * HD + t] * ns;       // noisy sample
        float d = lp * (xv - mean) * inv;       // (noisy_mean - mean)/ (std+eps)
        t2 += d * d;
        S2 += ln * ln;
    }
    noisy_emb[(size_t)g * HD + t] = nsum;

    // numerator contribution of feature j: sum_i term1 + counts * term2_j
    float r = stdv * stdv * inv * inv;
    s_red[t] = 0.5f * S2 * r + (float)NPG * t2;
    __syncthreads();

    // tree-reduce s_red over 128 features
#pragma unroll
    for (int off = 64; off > 0; off >>= 1) {
        if (t < off) s_red[t] += s_red[t + off];
        __syncthreads();
    }

    // ---------------- edges: 2x2 outer-product segment sum ------------------
    {
        int e = g * EPG + t;                 // 128 edges, 1 per thread
        int si = esrc[e] - g * NPG;
        int di = edst[e] - g * NPG;
        float a0 = s_p0[si], a1 = s_p1[si];
        float c0 = s_p0[di], c1 = s_p1[di];
        atomicAdd(&s_adj[0], a0 * c0);       // ds_add_f32
        atomicAdd(&s_adj[1], a0 * c1);
        atomicAdd(&s_adj[2], a1 * c0);
        atomicAdd(&s_adj[3], a1 * c1);
    }
    __syncthreads();

    // ---------------- scalar outputs (global float atomics) -----------------
    if (t == 0) {
        float invG = 1.0f / (float)G;
        // kl_loss
        atomicAdd(&scalars[1], s_red[0] * (1.0f / (NPG * HD)) * invG);
        // pos_penalty from normalized-adjacency diagonal
        float l1a = fmaxf(fabsf(s_adj[0]) + fabsf(s_adj[1]), 1e-12f);
        float l1b = fmaxf(fabsf(s_adj[2]) + fabsf(s_adj[3]), 1e-12f);
        float d0 = s_adj[0] / l1a - 1.0f;
        float d1 = s_adj[3] / l1b - 1.0f;
        atomicAdd(&scalars[0], 0.5f * (d0 * d0 + d1 * d1) * invG);
        // preserve_rate
        atomicAdd(&scalars[2], ((float)s_pres * (1.0f / NPG)) * invG);
    }
}

// ---------------------------------------------------------------------------
// Launch. Input order: x, W1, b1, W2, b2, gumbel_noise, uniform_noise,
//                      edge_index, batch, num_graphs
// Output order: graph_emb (G*H), noisy_emb (G*H), active (N),
//               pos_penalty, kl_loss, preserve_rate
// ---------------------------------------------------------------------------
extern "C" void kernel_launch(void* const* d_in, const int* in_sizes, int n_in,
                              void* d_out, int out_size, void* d_ws, size_t ws_size,
                              hipStream_t stream) {
    const float* x       = (const float*)d_in[0];
    const float* W1      = (const float*)d_in[1];
    const float* b1      = (const float*)d_in[2];
    const float* W2      = (const float*)d_in[3];
    const float* b2      = (const float*)d_in[4];
    const float* gumbel  = (const float*)d_in[5];
    const float* uniform = (const float*)d_in[6];
    const int*   eidx    = (const int*)d_in[7];
    // d_in[8] = batch, d_in[9] = num_graphs (device); derive G host-side:
    const int N = in_sizes[8];          // batch length = total nodes
    const int G = N / NPG;
    const int E = G * EPG;

    const int* esrc = eidx;             // edge_index[0]
    const int* edst = eidx + E;         // edge_index[1]

    float* out       = (float*)d_out;
    float* graph_emb = out;                         // G*HD
    float* noisy_emb = out + (size_t)G * HD;        // G*HD
    float* active    = out + (size_t)2 * G * HD;    // N
    float* scalars   = out + (size_t)2 * G * HD + N;// pos_penalty, kl, preserve

    _Float16* w1p = (_Float16*)d_ws;                // 32 KB packed f16 W1

    pack_w1_kernel<<<8, 128, 0, stream>>>(W1, w1p);
    zero_scalars_kernel<<<1, 32, 0, stream>>>(scalars);
    gib_fused_kernel<<<G, 128, 0, stream>>>(x, w1p, b1, W2, b2, gumbel, uniform,
                                            esrc, edst, graph_emb, noisy_emb,
                                            active, scalars, G);
}
